// GNNChildDecoder_30872224923906
// MI455X (gfx1250) — compile-verified, matrix operand order checked by reading, not run on hardware
//
#include <hip/hip_runtime.h>
#include <hip/hip_bf16.h>

typedef __attribute__((ext_vector_type(16))) __bf16 v16bf;
typedef __attribute__((ext_vector_type(8)))  float  v8f;

#define NEGBIG (-1e30f)

// ---------------------------------------------------------------------------
// K1: child = relu(x @ W_parent + b_parent); exists = child @ W_exists + b_ex
// one block per node m (256 blocks x 256 threads)
// ---------------------------------------------------------------------------
__global__ __launch_bounds__(256) void k_parent(
    const float* __restrict__ pf, const float* __restrict__ gcc,
    const float* __restrict__ gnc,
    const float* __restrict__ Wp, const float* __restrict__ bp,
    const float* __restrict__ Wex, const float* __restrict__ bex,
    float* __restrict__ cf0, float* __restrict__ exists_out,
    int* __restrict__ nodeok)
{
    __shared__ float x[282];
    __shared__ float red[256];
    int m = blockIdx.x, k = threadIdx.x;
    x[k] = pf[k];
    if (k < 16) x[256 + k] = gcc[k];
    if (k < 10) x[272 + k] = gnc[k];
    __syncthreads();
    int col = m * 256 + k;
    float acc = bp[col];
    for (int i = 0; i < 282; ++i)
        acc = fmaf(x[i], Wp[i * 65536 + col], acc);
    float c = fmaxf(acc, 0.f);
    cf0[col] = c;
    red[k] = c * Wex[k];
    __syncthreads();
    for (int s = 128; s > 0; s >>= 1) {
        if (k < s) red[k] += red[k + s];
        __syncthreads();
    }
    if (k == 0) {
        float e = red[0] + bex[0];
        exists_out[m] = e;
        nodeok[m] = (e > 0.f) ? 1 : 0;
    }
}

// ---------------------------------------------------------------------------
// Dual 256x256x256 GEMM: C1 = A@W1 (+bias1), C2 = A@W2. One block per row.
// used for (a' = cf0@Wel_a + b_el, b2 = cf0@Wel_b) and (As = cf@Ws + b_ne, Ad)
// ---------------------------------------------------------------------------
__global__ __launch_bounds__(256) void k_gemm_dual(
    const float* __restrict__ A, const float* __restrict__ W1,
    const float* __restrict__ W2, const float* __restrict__ bias1,
    float* __restrict__ C1, float* __restrict__ C2)
{
    __shared__ float row[256];
    int m = blockIdx.x, k = threadIdx.x;
    row[k] = A[m * 256 + k];
    __syncthreads();
    float a1 = bias1 ? bias1[k] : 0.f;
    float a2 = 0.f;
    for (int h = 0; h < 256; ++h) {
        float r = row[h];
        a1 = fmaf(r, W1[h * 256 + k], a1);
        a2 = fmaf(r, W2[h * 256 + k], a2);
    }
    C1[m * 256 + k] = a1;
    C2[m * 256 + k] = a2;
}

// ---------------------------------------------------------------------------
// K3: ee[m,n,t] = sum_h relu(a'[m,h]+b2[n,h]) * Wee[t,h] + bee[t]
//     mask bits + global has_edges flag. block = m, thread = n.
// ---------------------------------------------------------------------------
__global__ __launch_bounds__(256) void k_ee_mask(
    const float* __restrict__ aP, const float* __restrict__ b2,
    const float* __restrict__ Wee, const float* __restrict__ bee,
    const int* __restrict__ nodeok, float* __restrict__ ee_out,
    unsigned* __restrict__ mask32, int* __restrict__ hasEdges)
{
    __shared__ float am[256];
    __shared__ float wt[4][256];
    __shared__ unsigned red[256];
    int m = blockIdx.x, n = threadIdx.x;
    am[n] = aP[m * 256 + n];
    for (int t = 0; t < 4; ++t) wt[t][n] = Wee[t * 256 + n];
    __syncthreads();
    float a0 = bee[0], a1 = bee[1], a2 = bee[2], a3 = bee[3];
    const float* b2n = b2 + n * 256;
    for (int h = 0; h < 256; ++h) {
        float e = am[h] + b2n[h];
        e = e > 0.f ? e : 0.f;
        a0 = fmaf(e, wt[0][h], a0);
        a1 = fmaf(e, wt[1][h], a1);
        a2 = fmaf(e, wt[2][h], a2);
        a3 = fmaf(e, wt[3][h], a3);
    }
    int idx = (m * 256 + n) * 4;
    ee_out[idx] = a0; ee_out[idx + 1] = a1;
    ee_out[idx + 2] = a2; ee_out[idx + 3] = a3;
    unsigned bits = 0u;
    if (nodeok[m] && nodeok[n])
        bits = (a0 > 0.f ? 1u : 0u) | (a1 > 0.f ? 2u : 0u) |
               (a2 > 0.f ? 4u : 0u) | (a3 > 0.f ? 8u : 0u);
    mask32[m * 256 + n] = bits;
    red[n] = bits;
    __syncthreads();
    for (int s = 128; s > 0; s >>= 1) {
        if (n < s) red[n] |= red[n + s];
        __syncthreads();
    }
    if (n == 0 && red[0]) atomicOr(hasEdges, 1);
}

// ---------------------------------------------------------------------------
// Pack We (f32 [256h x 256k]) into WMMA B-fragment order, bf16.
// offset = (((kt*8 + s)*32) + lane)*16 + j ; j=2v+lh -> element (h,k)
// ---------------------------------------------------------------------------
__global__ __launch_bounds__(256) void k_pack_we(
    const float* __restrict__ We, __bf16* __restrict__ out)
{
    int g = blockIdx.x * 256 + threadIdx.x;     // 0..65535
    int j = g & 15;
    int lane = (g >> 4) & 31;
    int s = (g >> 9) & 7;
    int kt = g >> 12;
    int v = j >> 1, lh = j & 1;
    int col = kt * 16 + (lane & 15);
    int h = s * 32 + ((v & 4) ? 16 : 0) + (lane >> 4) * 8 + (v & 3) * 2 + lh;
    out[g] = (__bf16)We[h * 256 + col];
}

// ---------------------------------------------------------------------------
// K4: fused per-m kernel.  For one m:
//   el[n,h] = relu(a'[m,h]+b2[n,h])  (16-row LDS tile, regenerated per n-tile)
//   E = el @ We  via v_wmma_f32_16x16x32_bf16  (8 chained WMMAs, K=256)
//   pre = As'[m,k] + Ad[n,k] + E ; cand_t = relu(pre + ee[m,n,t]*Wt[t,k])
//   new[m,k] = max over masked (n,t);  cf_next = has_edges ? max(new,0) : cf
// 256 threads = 8 waves; wave w owns k-tiles 2w, 2w+1; loops 16 n-tiles.
// ---------------------------------------------------------------------------
__global__ __launch_bounds__(256) void k_fused(
    const float* __restrict__ aP, const float* __restrict__ b2,
    const float* __restrict__ As, const float* __restrict__ Ad,
    const float* __restrict__ Wt, const float* __restrict__ ee,
    const unsigned* __restrict__ mask32, const int* __restrict__ hasEdges,
    const __bf16* __restrict__ WeP,
    const float* __restrict__ cf_prev, float* __restrict__ cf_next)
{
    __shared__ float aPm[256];
    __shared__ float asr[256];
    __shared__ float wts[4][256];
    __shared__ float el[16 * 260];     // stride 260 avoids LDS bank conflicts
    __shared__ float eet[16][4];
    __shared__ unsigned mb[16];

    int m = blockIdx.x;
    int tid = threadIdx.x;
    int lane = tid & 31, wave = tid >> 5;
    aPm[tid] = aP[m * 256 + tid];
    asr[tid] = As[m * 256 + tid];     // already includes b_ne
    for (int t = 0; t < 4; ++t) wts[t][tid] = Wt[t * 256 + tid];
    int he = hasEdges[0];
    __syncthreads();

    float acc[2] = {NEGBIG, NEGBIG};
    int row = lane & 15, hf = lane >> 4;

    for (int nt = 0; nt < 16; ++nt) {
        // build el tile [16 x 256] cooperatively
        {
            int nl = tid >> 4;
            int n = nt * 16 + nl;
            int hb = (tid & 15) * 16;
            const float* b2n = b2 + n * 256 + hb;
            const float* ap = &aPm[hb];
            float* er = &el[nl * 260 + hb];
#pragma unroll
            for (int j = 0; j < 16; ++j) {
                float e = ap[j] + b2n[j];
                er[j] = e > 0.f ? e : 0.f;
            }
        }
        if (tid < 64) {
            int nl = tid >> 2, t = tid & 3;
            eet[nl][t] = ee[(m * 256 + nt * 16 + nl) * 4 + t];
        }
        if (tid < 16) mb[tid] = mask32[m * 256 + nt * 16 + tid];
        __syncthreads();

        // A fragments (bf16) from LDS, shared by both k-tiles of this wave
        v16bf af[8];
#pragma unroll
        for (int s = 0; s < 8; ++s) {
#pragma unroll
            for (int v = 0; v < 8; ++v) {
                int kb = s * 32 + ((v & 4) ? 16 : 0) + hf * 8 + (v & 3) * 2;
                const float* p = &el[row * 260 + kb];
                af[s][2 * v] = (__bf16)p[0];
                af[s][2 * v + 1] = (__bf16)p[1];
            }
        }
#pragma unroll
        for (int q = 0; q < 2; ++q) {
            int kt = wave * 2 + q;
            v8f c = {0.f, 0.f, 0.f, 0.f, 0.f, 0.f, 0.f, 0.f};
#pragma unroll
            for (int s = 0; s < 8; ++s) {
                v16bf bf = *(const v16bf*)(WeP + (((kt * 8 + s) * 32) + lane) * 16);
                c = __builtin_amdgcn_wmma_f32_16x16x32_bf16(
                        false, af[s], false, bf, (short)0, c, false, false);
            }
            // epilogue: masked max over rows (n) and edge types (t)
            int kc = kt * 16 + (lane & 15);
            float ask = asr[kc];
            float w0 = wts[0][kc], w1 = wts[1][kc], w2 = wts[2][kc], w3 = wts[3][kc];
            float a = acc[q];
#pragma unroll
            for (int v = 0; v < 8; ++v) {
                int nl = v + hf * 8;
                unsigned bits = mb[nl];
                if (bits) {
                    int n = nt * 16 + nl;
                    float pre = ask + Ad[n * 256 + kc] + c[v];
                    if (bits & 1u) { float cd = pre + eet[nl][0] * w0; cd = cd > 0.f ? cd : 0.f; a = a > cd ? a : cd; }
                    if (bits & 2u) { float cd = pre + eet[nl][1] * w1; cd = cd > 0.f ? cd : 0.f; a = a > cd ? a : cd; }
                    if (bits & 4u) { float cd = pre + eet[nl][2] * w2; cd = cd > 0.f ? cd : 0.f; a = a > cd ? a : cd; }
                    if (bits & 8u) { float cd = pre + eet[nl][3] * w3; cd = cd > 0.f ? cd : 0.f; a = a > cd ? a : cd; }
                }
            }
            acc[q] = a;
        }
        __syncthreads();
    }
    // merge the two lane halves (rows n+0..7 vs n+8..15 share the same k col)
    acc[0] = fmaxf(acc[0], __shfl_xor(acc[0], 16, 32));
    acc[1] = fmaxf(acc[1], __shfl_xor(acc[1], 16, 32));
    if (lane < 16) {
#pragma unroll
        for (int q = 0; q < 2; ++q) {
            int kc = (wave * 2 + q) * 16 + lane;
            float nv = fmaxf(acc[q], 0.f);
            cf_next[m * 256 + kc] = he ? nv : cf_prev[m * 256 + kc];
        }
    }
}

// ---------------------------------------------------------------------------
// K5: ch = relu([cf0|cf1|cf2] @ W_child + b); feats = relu(ch@Wc2+b); sem head
// ---------------------------------------------------------------------------
__global__ __launch_bounds__(256) void k_child(
    const float* __restrict__ cf0, const float* __restrict__ cf1,
    const float* __restrict__ cf2,
    const float* __restrict__ Wc, const float* __restrict__ bc,
    const float* __restrict__ Wc2, const float* __restrict__ bc2,
    const float* __restrict__ Wsem, const float* __restrict__ bsem,
    float* __restrict__ out_feats, float* __restrict__ out_sem)
{
    __shared__ float row[768];
    __shared__ float ch[256];
    int m = blockIdx.x, k = threadIdx.x;
    row[k] = cf0[m * 256 + k];
    row[256 + k] = cf1[m * 256 + k];
    row[512 + k] = cf2[m * 256 + k];
    __syncthreads();
    float a = bc[k];
    for (int h = 0; h < 768; ++h) a = fmaf(row[h], Wc[h * 256 + k], a);
    ch[k] = fmaxf(a, 0.f);
    __syncthreads();
    float a2 = bc2[k];
    for (int h = 0; h < 256; ++h) a2 = fmaf(ch[h], Wc2[h * 256 + k], a2);
    out_feats[m * 256 + k] = fmaxf(a2, 0.f);
    if (k < 50) {
        float a3 = bsem[k];
        for (int h = 0; h < 256; ++h) a3 = fmaf(ch[h], Wsem[h * 50 + k], a3);
        out_sem[m * 50 + k] = a3;
    }
}

// ---------------------------------------------------------------------------
extern "C" void kernel_launch(void* const* d_in, const int* in_sizes, int n_in,
                              void* d_out, int out_size, void* d_ws, size_t ws_size,
                              hipStream_t stream)
{
    (void)in_sizes; (void)n_in; (void)out_size; (void)ws_size;
    const float* pf   = (const float*)d_in[0];
    const float* gcc  = (const float*)d_in[1];
    const float* gnc  = (const float*)d_in[2];
    const float* Wp   = (const float*)d_in[3];
    const float* bp   = (const float*)d_in[4];
    const float* Wex  = (const float*)d_in[5];
    const float* bex  = (const float*)d_in[6];
    const float* Wsem = (const float*)d_in[7];
    const float* bsem = (const float*)d_in[8];
    const float* Wel  = (const float*)d_in[9];
    const float* bel  = (const float*)d_in[10];
    const float* Wee  = (const float*)d_in[11];
    const float* bee  = (const float*)d_in[12];
    const float* Wne  = (const float*)d_in[13];
    const float* bne  = (const float*)d_in[14];
    const float* Wc   = (const float*)d_in[15];
    const float* bc   = (const float*)d_in[16];
    const float* Wc2  = (const float*)d_in[17];
    const float* bc2  = (const float*)d_in[18];

    // output layout (tuple concat, flat f32):
    // child_feats[65536] | sem[12800] | exists[256] | ee[262144]
    float* out = (float*)d_out;
    float* out_feats = out;
    float* out_sem   = out + 65536;
    float* out_exist = out + 78336;
    float* out_ee    = out + 78592;

    // workspace layout
    float* ws  = (float*)d_ws;
    float* cf0 = ws;
    float* cf1 = cf0 + 65536;
    float* cf2 = cf1 + 65536;
    float* aP  = cf2 + 65536;          // a + b_el
    float* b2  = aP + 65536;
    float* As  = b2 + 65536;           // cf@Ws + b_ne
    float* Ad  = As + 65536;           // cf@Wd
    unsigned* mask32 = (unsigned*)(Ad + 65536);   // 65536 u32
    int* nodeok   = (int*)(mask32 + 65536);       // 256
    int* hasEdges = nodeok + 256;                 // 1 (+pad)
    __bf16* WeP   = (__bf16*)(hasEdges + 64);     // 65536 bf16, fragment-packed

    hipMemsetAsync(hasEdges, 0, sizeof(int), stream);

    k_parent<<<256, 256, 0, stream>>>(pf, gcc, gnc, Wp, bp, Wex, bex,
                                      cf0, out_exist, nodeok);
    // a' = cf0 @ Wel[:256] + b_el ; b2 = cf0 @ Wel[256:]
    k_gemm_dual<<<256, 256, 0, stream>>>(cf0, Wel, Wel + 65536, bel, aP, b2);
    k_ee_mask<<<256, 256, 0, stream>>>(aP, b2, Wee, bee, nodeok,
                                       out_ee, mask32, hasEdges);

    float* cfs[3] = {cf0, cf1, cf2};
    for (int i = 0; i < 2; ++i) {
        const float* Wni = Wne + i * (772 * 256);
        k_pack_we<<<256, 256, 0, stream>>>(Wni + 512 * 256, WeP);
        k_gemm_dual<<<256, 256, 0, stream>>>(cfs[i], Wni, Wni + 65536,
                                             bne + i * 256, As, Ad);
        k_fused<<<256, 256, 0, stream>>>(aP, b2, As, Ad, Wni + 768 * 256,
                                         out_ee, mask32, hasEdges, WeP,
                                         cfs[i], cfs[i + 1]);
    }
    k_child<<<256, 256, 0, stream>>>(cf0, cf1, cf2, Wc, bc, Wc2, bc2,
                                     Wsem, bsem, out_feats, out_sem);
}
